// Attention_88519275970960
// MI455X (gfx1250) — matrix operand order repaired and checked by reading, not compiled, not run on hardware
//
#include <hip/hip_runtime.h>
#include <hip/hip_bf16.h>

// ---------------------------------------------------------------------------
// Problem constants (match reference): B=8, T=1024, C=512, H=8, D=64, TT=4
// ---------------------------------------------------------------------------
#define BB 8
#define TT_ 4
#define TSEQ 1024
#define CE 512
#define NH 8
#define DH 64
#define C3 (3 * CE)

typedef __bf16 bf16_t;
typedef __attribute__((ext_vector_type(16))) __bf16 v16bf;
typedef __attribute__((ext_vector_type(8)))  float  v8f;
typedef int v4i_b __attribute__((vector_size(16)));  // matches builtin param

union FragBF { v16bf v; uint4 q[2]; };

// ---------------------------------------------------------------------------
// CDNA5 async global->LDS copy (ASYNCcnt path), with synchronous fallback
// ---------------------------------------------------------------------------
#if __has_builtin(__builtin_amdgcn_global_load_async_to_lds_b128)
#define ASYNC_LDS 1
#else
#define ASYNC_LDS 0
#endif

__device__ __forceinline__ void copy16_g2l(bf16_t* ldst, const bf16_t* gsrc) {
#if ASYNC_LDS
  __builtin_amdgcn_global_load_async_to_lds_b128(
      (__attribute__((address_space(1))) v4i_b*)gsrc,
      (__attribute__((address_space(3))) v4i_b*)ldst, 0, 0);
#else
  *reinterpret_cast<uint4*>(ldst) = *reinterpret_cast<const uint4*>(gsrc);
#endif
}

template <int N>
__device__ __forceinline__ void wait_async() {
#if ASYNC_LDS
#if __has_builtin(__builtin_amdgcn_s_wait_asynccnt)
  __builtin_amdgcn_s_wait_asynccnt(N);
#else
  asm volatile("s_wait_asynccnt %0" ::"i"(N) : "memory");
#endif
#endif
}

__device__ __forceinline__ v8f wmma_bf16(v16bf a, v16bf b, v8f c) {
  return __builtin_amdgcn_wmma_f32_16x16x32_bf16(
      /*neg_a=*/false, a, /*neg_b=*/false, b,
      /*c_mod=*/(short)0, c, /*reuse_a=*/false, /*reuse_b=*/false);
}

// A fragment (16x32, bf16): matrix stored row-major with leading dim ld.
__device__ __forceinline__ v16bf load_a(const bf16_t* base, int ld, int rb,
                                        int kb, int lane) {
  int r  = rb + (lane & 15);
  int h8 = (lane >> 4) * 8;
  const bf16_t* p = base + (size_t)r * ld + kb + h8;
  FragBF f;
  f.q[0] = *reinterpret_cast<const uint4*>(p);       // K = h8 + 0..7
  f.q[1] = *reinterpret_cast<const uint4*>(p + 16);  // K = h8 + 16..23
  return f.v;
}

// B fragment (32x16, bf16): matrix stored [N][K] (K contiguous), leading dim ld.
__device__ __forceinline__ v16bf load_b(const bf16_t* base, int ld, int nb,
                                        int kb, int lane) {
  int n   = nb + (lane & 15);
  int h16 = (lane >> 4) * 8;
  const bf16_t* p = base + (size_t)n * ld + kb + h16;
  FragBF f;
  f.q[0] = *reinterpret_cast<const uint4*>(p);       // K = h8 + 0..7
  f.q[1] = *reinterpret_cast<const uint4*>(p + 16);  // K = h8 + 16..23
  return f.v;
}

// ---------------------------------------------------------------------------
// fp32 -> bf16 conversion (float4 loads, 8-byte packed stores)
// ---------------------------------------------------------------------------
__global__ __launch_bounds__(256) void cvt_f32_bf16(const float* __restrict__ in,
                                                    bf16_t* __restrict__ out,
                                                    int n4) {
  int i = blockIdx.x * blockDim.x + threadIdx.x;
  if (i >= n4) return;
  float4 f = reinterpret_cast<const float4*>(in)[i];
  union { bf16_t h[4]; uint2 u; } r;
  r.h[0] = (bf16_t)f.x; r.h[1] = (bf16_t)f.y;
  r.h[2] = (bf16_t)f.z; r.h[3] = (bf16_t)f.w;
  reinterpret_cast<uint2*>(out)[i] = r.u;
}

// ---------------------------------------------------------------------------
// Pipelined WMMA GEMM: C[M,N] = A[M,K] bf16 * B[K,N] bf16 + bias
// Block tile 128x64, BK=64, double-buffered LDS, async A-tile copies.
// 8 waves, each computing a 32x32 quadrant (2x2 WMMA tiles, 2 K-steps/phase).
// ---------------------------------------------------------------------------
template <bool OUT_BF16>
__global__ __launch_bounds__(256) void gemm_bf16(
    const bf16_t* __restrict__ A, const bf16_t* __restrict__ Bm,
    const float* __restrict__ bias, void* __restrict__ Cout,
    int M, int N, int K) {
  constexpr int BM = 128, BN = 64, BK = 64, PAD = 8, LD = BK + PAD;
  __shared__ alignas(16) bf16_t As[2][BM][LD];  // row-major [m][k]
  __shared__ alignas(16) bf16_t Bs[2][BN][LD];  // transposed [n][k]

  const int tid  = threadIdx.x;
  const int lane = tid & 31;
  const int wave = tid >> 5;
  const int wm   = wave & 3;   // 4 waves along M
  const int wn   = wave >> 2;  // 2 waves along N
  const int bm   = blockIdx.y * BM;
  const int bn   = blockIdx.x * BN;
  const int NT   = K / BK;
  constexpr int A_ISSUES = (BM * BK / 8) / 256;  // 4 async copies / thread / tile

  uint4 breg[2];  // staged B tile (512 uint4 / 256 threads)

  auto issueA = [&](int buf, int t) {
#pragma unroll
    for (int j = 0; j < A_ISSUES; ++j) {
      int i = tid + j * 256;
      int r = i >> 3, c = (i & 7) * 8;
      copy16_g2l(&As[buf][r][c],
                 &A[(size_t)(bm + r) * K + t * BK + c]);
    }
  };
  auto loadB = [&](int t) {
#pragma unroll
    for (int j = 0; j < 2; ++j) {
      int i = tid + j * 256;
      int kk = i >> 3, c = (i & 7) * 8;
      breg[j] = *reinterpret_cast<const uint4*>(
          &Bm[(size_t)(t * BK + kk) * N + bn + c]);
    }
  };
  auto scatterB = [&](int buf) {
#pragma unroll
    for (int j = 0; j < 2; ++j) {
      int i = tid + j * 256;
      int kk = i >> 3, c = (i & 7) * 8;
      const bf16_t* tp = reinterpret_cast<const bf16_t*>(&breg[j]);
#pragma unroll
      for (int e = 0; e < 8; ++e) Bs[buf][c + e][kk] = tp[e];
    }
  };

  v8f acc[2][2] = {};
  issueA(0, 0);
  loadB(0);

  for (int t = 0; t < NT; ++t) {
    const int cur = t & 1, nxt = cur ^ 1;
    scatterB(cur);                       // tile t's B (regs -> LDS)
    const bool more = (t + 1 < NT);
    if (more) {
      issueA(nxt, t + 1);                // next A tile: async, stays in flight
      loadB(t + 1);                      // next B tile: global -> regs
    }
    if (more) wait_async<A_ISSUES>(); else wait_async<0>();
    __syncthreads();                     // tile t fully resident

#pragma unroll
    for (int ks = 0; ks < 2; ++ks) {
      v16bf af[2], bfr[2];
      af[0]  = load_a(&As[cur][0][0], LD, wm * 32 + 0,  ks * 32, lane);
      af[1]  = load_a(&As[cur][0][0], LD, wm * 32 + 16, ks * 32, lane);
      bfr[0] = load_b(&Bs[cur][0][0], LD, wn * 32 + 0,  ks * 32, lane);
      bfr[1] = load_b(&Bs[cur][0][0], LD, wn * 32 + 16, ks * 32, lane);
#pragma unroll
      for (int mi = 0; mi < 2; ++mi)
#pragma unroll
        for (int ni = 0; ni < 2; ++ni)
          acc[mi][ni] = wmma_bf16(af[mi], bfr[ni], acc[mi][ni]);
    }
    __syncthreads();                     // done reading cur before reuse
  }

  // Epilogue: C layout — lane&15 = column, vgpr e -> row e + 8*(lane>>4)
  const int nlo  = lane & 15;
  const int half = lane >> 4;
#pragma unroll
  for (int mi = 0; mi < 2; ++mi)
#pragma unroll
    for (int ni = 0; ni < 2; ++ni) {
      int n = bn + wn * 32 + ni * 16 + nlo;
      float bv = bias[n];
#pragma unroll
      for (int e = 0; e < 8; ++e) {
        int m = bm + wm * 32 + mi * 16 + e + half * 8;
        float v = acc[mi][ni][e] + bv;
        if (OUT_BF16)
          reinterpret_cast<bf16_t*>(Cout)[(size_t)m * N + n] = (bf16_t)v;
        else
          reinterpret_cast<float*>(Cout)[(size_t)m * N + n] = v;
      }
    }
}

// ---------------------------------------------------------------------------
// Pipelined flash attention: one workgroup (4 waves) per (b*H+h, 64-q block).
// Wave w owns query rows [qb*64 + 16w, +16). S = Q K^T via WMMA, online
// softmax in f32, P V via WMMA. Mask: (k <= q) || padding_mask[b%TT][q].
// K tiles: async double-buffered; V tiles: register-staged transpose.
// ---------------------------------------------------------------------------
__global__ __launch_bounds__(128) void attn_kernel(
    const bf16_t* __restrict__ qkv,           // [B,T,3C] bf16
    const unsigned char* __restrict__ pmask,  // [TT,T] bool
    bf16_t* __restrict__ y) {                 // [B,T,C] bf16
  constexpr int BKV = 64, PAD = 8, LDK = DH + PAD, LDV = BKV + PAD;
  __shared__ alignas(16) bf16_t Ks[2][BKV][LDK];  // [kv][d]
  __shared__ alignas(16) bf16_t Vs[2][DH][LDV];   // [d][kv] (transposed)
  __shared__ alignas(16) bf16_t Ps[4][16][LDV];   // per-wave P scratch

  const int tid  = threadIdx.x;
  const int lane = tid & 31;
  const int wave = tid >> 5;
  const int nlo  = lane & 15;
  const int half = lane >> 4;

  const int qb = blockIdx.x;
  const int bh = blockIdx.y;
  const int b  = bh >> 3;
  const int h  = bh & 7;
  const int tt = b & (TT_ - 1);
  const int q0 = qb * 64 + wave * 16;

  // Q fragments held in registers for the whole kernel
  const bf16_t* qbase = qkv + ((size_t)b * TSEQ + q0) * C3 + h * DH;
  v16bf qf[2];
  qf[0] = load_a(qbase, C3, 0, 0,  lane);
  qf[1] = load_a(qbase, C3, 0, 32, lane);

  int   qrow[8];
  float padq[8], rmax[8], rsum[8];
#pragma unroll
  for (int e = 0; e < 8; ++e) {
    qrow[e] = q0 + e + 8 * half;
    padq[e] = pmask[(size_t)tt * TSEQ + qrow[e]] ? 1.f : 0.f;
    rmax[e] = -1e30f;
    rsum[e] = 0.f;
  }
  v8f oacc[4] = {};

  constexpr int K_ISSUES = (BKV * DH / 8) / 128;  // 4 async copies / thread
  uint4 vreg[4];

  auto issueK = [&](int buf, int kv0) {
#pragma unroll
    for (int j = 0; j < K_ISSUES; ++j) {
      int i = tid + j * 128;
      int kr = i >> 3, c = (i & 7) * 8;
      copy16_g2l(&Ks[buf][kr][c],
                 &qkv[((size_t)b * TSEQ + kv0 + kr) * C3 + CE + h * DH + c]);
    }
  };
  auto loadV = [&](int kv0) {
#pragma unroll
    for (int j = 0; j < 4; ++j) {
      int i = tid + j * 128;
      int kr = i >> 3, c = (i & 7) * 8;
      vreg[j] = *reinterpret_cast<const uint4*>(
          &qkv[((size_t)b * TSEQ + kv0 + kr) * C3 + 2 * CE + h * DH + c]);
    }
  };
  auto scatterV = [&](int buf) {
#pragma unroll
    for (int j = 0; j < 4; ++j) {
      int i = tid + j * 128;
      int kr = i >> 3, c = (i & 7) * 8;
      const bf16_t* tp = reinterpret_cast<const bf16_t*>(&vreg[j]);
#pragma unroll
      for (int e = 0; e < 8; ++e) Vs[buf][c + e][kr] = tp[e];
    }
  };

  issueK(0, 0);
  loadV(0);

#pragma unroll 1
  for (int t = 0; t < TSEQ / BKV; ++t) {
    const int cur = t & 1, nxt = cur ^ 1;
    const int kv0 = t * BKV;
    scatterV(cur);
    const bool more = (t + 1 < TSEQ / BKV);
    if (more) {
      issueK(nxt, kv0 + BKV);
      loadV(kv0 + BKV);
    }
    if (more) wait_async<K_ISSUES>(); else wait_async<0>();
    __syncthreads();

    // S = Q K^T : 16 x 64 (four 16x16 tiles, K-dim 64 = 2 WMMA steps)
    v8f s[4] = {};
#pragma unroll
    for (int ni = 0; ni < 4; ++ni) {
      v16bf kf0 = load_b(&Ks[cur][0][0], LDK, ni * 16, 0,  lane);
      v16bf kf1 = load_b(&Ks[cur][0][0], LDK, ni * 16, 32, lane);
      s[ni] = wmma_bf16(qf[0], kf0, s[ni]);
      s[ni] = wmma_bf16(qf[1], kf1, s[ni]);
    }

    // scale + mask + block row-max
    float mblk[8];
#pragma unroll
    for (int e = 0; e < 8; ++e) mblk[e] = -1e30f;
#pragma unroll
    for (int ni = 0; ni < 4; ++ni) {
      int k = kv0 + ni * 16 + nlo;
#pragma unroll
      for (int e = 0; e < 8; ++e) {
        bool ok  = (k <= qrow[e]) || (padq[e] != 0.f);
        float sv = ok ? s[ni][e] * 0.125f : -1e30f;
        s[ni][e] = sv;
        mblk[e]  = fmaxf(mblk[e], sv);
      }
    }
#pragma unroll
    for (int e = 0; e < 8; ++e) {
      float m = mblk[e];
      m = fmaxf(m, __shfl_xor(m, 1, 16));
      m = fmaxf(m, __shfl_xor(m, 2, 16));
      m = fmaxf(m, __shfl_xor(m, 4, 16));
      m = fmaxf(m, __shfl_xor(m, 8, 16));
      mblk[e] = m;
    }

    float alpha[8], ls[8];
#pragma unroll
    for (int e = 0; e < 8; ++e) {
      float mnew = fmaxf(rmax[e], mblk[e]);
      alpha[e] = __expf(rmax[e] - mnew);
      rmax[e]  = mnew;
      ls[e]    = 0.f;
    }
#pragma unroll
    for (int ni = 0; ni < 4; ++ni)
#pragma unroll
      for (int e = 0; e < 8; ++e) {
        float p = __expf(s[ni][e] - rmax[e]);
        s[ni][e] = p;
        ls[e] += p;
      }
#pragma unroll
    for (int e = 0; e < 8; ++e) {
      float tsum = ls[e];
      tsum += __shfl_xor(tsum, 1, 16);
      tsum += __shfl_xor(tsum, 2, 16);
      tsum += __shfl_xor(tsum, 4, 16);
      tsum += __shfl_xor(tsum, 8, 16);
      rsum[e] = rsum[e] * alpha[e] + tsum;
    }
#pragma unroll
    for (int di = 0; di < 4; ++di)
#pragma unroll
      for (int e = 0; e < 8; ++e) oacc[di][e] *= alpha[e];

    // P (C layout) -> per-wave LDS slab -> A fragments
#pragma unroll
    for (int ni = 0; ni < 4; ++ni)
#pragma unroll
      for (int e = 0; e < 8; ++e)
        Ps[wave][e + 8 * half][ni * 16 + nlo] = (bf16_t)s[ni][e];

    v16bf pf0 = load_a(&Ps[wave][0][0], LDV, 0, 0,  lane);
    v16bf pf1 = load_a(&Ps[wave][0][0], LDV, 0, 32, lane);
#pragma unroll
    for (int di = 0; di < 4; ++di) {
      v16bf vf0 = load_b(&Vs[cur][0][0], LDV, di * 16, 0,  lane);
      v16bf vf1 = load_b(&Vs[cur][0][0], LDV, di * 16, 32, lane);
      oacc[di] = wmma_bf16(pf0, vf0, oacc[di]);
      oacc[di] = wmma_bf16(pf1, vf1, oacc[di]);
    }
    __syncthreads();
  }

  // normalize and write y (bf16)
#pragma unroll
  for (int di = 0; di < 4; ++di)
#pragma unroll
    for (int e = 0; e < 8; ++e) {
      float ov = oacc[di][e] / rsum[e];
      y[((size_t)b * TSEQ + qrow[e]) * CE + h * DH + di * 16 + nlo] = (bf16_t)ov;
    }
}

// ---------------------------------------------------------------------------
// Launcher
// ---------------------------------------------------------------------------
extern "C" void kernel_launch(void* const* d_in, const int* in_sizes, int n_in,
                              void* d_out, int out_size, void* d_ws,
                              size_t ws_size, hipStream_t stream) {
  (void)in_sizes; (void)n_in; (void)out_size; (void)ws_size;
  const float* x      = (const float*)d_in[0];
  const unsigned char* pmask = (const unsigned char*)d_in[1];  // bool [TT,T]
  const float* Wqkv   = (const float*)d_in[2];
  const float* bqkv   = (const float*)d_in[3];
  const float* Wproj  = (const float*)d_in[4];
  const float* bproj  = (const float*)d_in[5];
  float* out = (float*)d_out;

  // workspace layout (bf16 buffers), ~44 MB total
  char* ws = (char*)d_ws;
  bf16_t* xb    = (bf16_t*)(ws + 0);                                     //  8 MB
  bf16_t* wqkvb = (bf16_t*)(ws + (size_t)8 * 1024 * 1024 + 512 * 1024);  // 1.5 MB
  bf16_t* wprjb = (bf16_t*)(ws + (size_t)10 * 1024 * 1024);              // 0.5 MB
  bf16_t* qkvb  = (bf16_t*)(ws + (size_t)11 * 1024 * 1024);              // 24 MB
  bf16_t* yb    = (bf16_t*)(ws + (size_t)36 * 1024 * 1024);              //  8 MB

  const int nx4 = (BB * TSEQ * CE) / 4;
  const int nw4 = (CE * C3) / 4;
  const int np4 = (CE * CE) / 4;
  cvt_f32_bf16<<<(nx4 + 255) / 256, 256, 0, stream>>>(x, xb, nx4);
  cvt_f32_bf16<<<(nw4 + 255) / 256, 256, 0, stream>>>(Wqkv, wqkvb, nw4);
  cvt_f32_bf16<<<(np4 + 255) / 256, 256, 0, stream>>>(Wproj, wprjb, np4);

  // qkv = x @ W_qkv + b_qkv   [8192,1536] bf16
  gemm_bf16<true><<<dim3(C3 / 64, (BB * TSEQ) / 128), 256, 0, stream>>>(
      xb, wqkvb, bqkv, qkvb, BB * TSEQ, C3, CE);

  // attention -> yb [8192,512] bf16
  attn_kernel<<<dim3(TSEQ / 64, BB * NH), 128, 0, stream>>>(qkvb, pmask, yb);

  // out = yb @ W_proj + b_proj   [8192,512] f32
  gemm_bf16<false><<<dim3(CE / 64, (BB * TSEQ) / 128), 256, 0, stream>>>(
      yb, wprjb, bproj, out, BB * TSEQ, CE, CE);
}
